// Decoder_15882789061187
// MI455X (gfx1250) — matrix operand order repaired and checked by reading, not compiled, not run on hardware
//
#include <hip/hip_runtime.h>
#include <hip/hip_bf16.h>
#include <math.h>

typedef _Float16 v16h __attribute__((ext_vector_type(16)));
typedef _Float16 h8   __attribute__((ext_vector_type(8)));
typedef float    v8f  __attribute__((ext_vector_type(8)));

#define NPTS    131072
#define MTILE   64
#define NBLK    (NPTS / MTILE)
#define THREADS 256

// ---- workspace layout (f16 units) ----
// All WMMA layers padded to Fp = 512 (nF = 32); layer 3's ftiles 16..31 are zero.
__constant__ int      c_nK[8]   = {9,16,16,16,17,16,16,16};   // Kp/32
__constant__ int      c_fout[8] = {512,512,512,253,512,512,512,512};
__constant__ unsigned c_wOff[8] = {0u,147456u,409600u,671744u,933888u,1212416u,1474560u,1736704u};
#define OFF8          1998848u
#define WS_HALF_COUNT 2007040u

// prep-side constants for all 9 layers
__constant__ int      p_fin[9]  = {259,512,512,512,512,512,512,512,512};
__constant__ int      p_fout[9] = {512,512,512,253,512,512,512,512,1};
__constant__ int      p_nK[9]   = {9,16,16,16,17,16,16,16,16};
__constant__ unsigned p_wOff[9] = {0u,147456u,409600u,671744u,933888u,1212416u,1474560u,1736704u,1998848u};

struct PrepParams {
    const float* W[9];
    const float* c[9];
};

struct DecParams {
    const float* b[9];
    const float* g[8];
    const float* be[8];
};

__device__ __forceinline__ float fast_tanh(float x) {
    // tanh(x) = sign(x) * (1 - e) / (1 + e), e = exp(-2|x|); stable for all x
    float ax = fabsf(x);
    float e  = __expf(-2.f * ax);
    float t  = (1.f - e) / (1.f + e);
    return copysignf(t, x);
}

// ---------------- zero workspace ----------------
__global__ void zero_ws_kernel(unsigned* __restrict__ ws32, unsigned nwords) {
    unsigned i = blockIdx.x * blockDim.x + threadIdx.x;
    unsigned stride = gridDim.x * blockDim.x;
    for (; i < nwords; i += stride) ws32[i] = 0u;
}

// ---------------- weight prep: Lipschitz row scaling + f16 B-fragment packing ----------------
__global__ __launch_bounds__(256) void prep_weights_kernel(PrepParams P, _Float16* __restrict__ ws) {
    const int layer = blockIdx.x;          // 0..8
    const int fin   = p_fin[layer];
    const int fout  = p_fout[layer];
    const int nK    = p_nK[layer];
    const unsigned base = p_wOff[layer];
    const float* W = P.W[layer];
    const float  cv = P.c[layer][0];
    const float  sp = (cv > 20.f) ? cv : log1pf(expf(cv));   // softplus(c)

    const int wave = threadIdx.x >> 5;
    const int lane = threadIdx.x & 31;

    for (int f = wave; f < fout; f += 8) {
        // abs row-sum across the wave
        float s = 0.f;
        for (int k = lane; k < fin; k += 32) s += fabsf(W[(size_t)f * fin + k]);
        #pragma unroll
        for (int off = 16; off > 0; off >>= 1) s += __shfl_xor(s, off, 32);
        const float scale = fminf(1.f, sp / s);

        for (int k0 = lane; k0 < fin; k0 += 32) {
            float wv = W[(size_t)f * fin + k0] * scale;
            int k = k0;
            if (layer == 0 && k0 >= 256) wv *= 100.f;                 // fold xyz*100 into W0
            if (layer == 4) k = (k0 < 253) ? k0 : (256 + (k0 - 253)); // latent re-injection reorder
            size_t dst;
            if (layer == 8) {
                dst = OFF8 + (unsigned)k;                             // plain vector
            } else {
                // B fragment in this code's contiguous K order (same as A): lanes 0-15 hold K 0-15, lanes 16-31 hold K 16-31,
                // 16 contiguous halves per lane per fragment
                int kt = k >> 5, kin = k & 31, ftile = f >> 4, n = f & 15;
                int lanep = n + ((kin >= 16) ? 16 : 0);
                int h = kin & 15;
                dst = (size_t)base + ((size_t)(ftile * nK + kt) << 9) + (size_t)lanep * 16 + h;
            }
            ws[dst] = (_Float16)wv;
        }
    }
}

// ---------------- uniform GEMM segment: acc += A(seg) x B(seg) ----------------
// Each wave computes 4 M-tiles x 4 F-tiles; trip counts/branches are wave-uniform.
// B addressed as uniform 64-bit base + 32-bit unsigned offset so the backend can
// use the GVS form (global_load_b128 v, v_off, s[base]) with no v_add_u64 per load.
__device__ __forceinline__ void gemm_seg(const _Float16* asrc, int astr,
                                         int kt0, int nKt,
                                         const char* __restrict__ wsb, unsigned wOffB, int nK,
                                         int wave, int ml, int kbA, int lane,
                                         v8f (&acc)[4][4]) {
    unsigned boff[4];
    #pragma unroll
    for (int ft = 0; ft < 4; ++ft)
        boff[ft] = wOffB + ((unsigned)((wave * 4 + ft) * nK + kt0) << 10) + (unsigned)lane * 32u;

    const _Float16* ap = asrc + (size_t)ml * astr + kt0 * 32 + kbA;

    for (int kt = 0; kt < nKt; ++kt) {
        v16h a[4];
        #pragma unroll
        for (int mt = 0; mt < 4; ++mt) {
            const _Float16* p = ap + mt * 16 * astr;
            h8 lo = *(const h8*)p;
            h8 hi = *(const h8*)(p + 8);
            a[mt] = __builtin_shufflevector(lo, hi, 0,1,2,3,4,5,6,7,8,9,10,11,12,13,14,15);
        }
        ap += 32;
        const unsigned kstep = (unsigned)kt << 10;
        #pragma unroll
        for (int ft = 0; ft < 4; ++ft) {
            const char* bpc = wsb + (boff[ft] + kstep);
            __builtin_prefetch((const void*)(bpc + 1024), 0, 3);   // next K-step fragment
            h8 blo = *(const h8*)bpc;
            h8 bhi = *(const h8*)(bpc + 16);
            v16h b = __builtin_shufflevector(blo, bhi, 0,1,2,3,4,5,6,7,8,9,10,11,12,13,14,15);
            #pragma unroll
            for (int mt = 0; mt < 4; ++mt)
                acc[mt][ft] = __builtin_amdgcn_wmma_f32_16x16x32_f16(
                    false, a[mt], false, b, (short)0, acc[mt][ft], false, false);
        }
    }
}

// ---------------- fused 9-layer decoder ----------------
// Single in-place activation buffer: each layer's output lives in f32 accumulators,
// so sAct can be overwritten after a barrier. LDS total = 100 KB -> 3 WGs per WGP.
__global__ __launch_bounds__(THREADS) void decoder_fused_kernel(
        DecParams P,
        const float* __restrict__ in,
        const _Float16* __restrict__ ws,
        float* __restrict__ out) {
    extern __shared__ _Float16 smem[];
    _Float16* sInp = smem;                     // [64][288] raw input (f16, zero padded)
    _Float16* sAct = smem + MTILE * 288;       // [64][512] activations (in-place)

    const int tid  = threadIdx.x;
    const int m0   = blockIdx.x * MTILE;
    const int wave = tid >> 5;
    const int lane = tid & 31;
    const int ml   = lane & 15;                // M (A) / N (B,C) lane coord
    const int kbA  = (lane < 16) ? 0 : 16;     // A-fragment K half base (contiguous 16)
    const int hiRow = (lane >> 4) << 3;        // C/D row offset (0 or 8)
    const char* wsb = (const char*)ws;

    // stage input tile, f32 -> f16, pad K 259..287 with zeros
    for (int i = tid; i < MTILE * 288; i += THREADS) {
        int m = i / 288, k = i - m * 288;
        float v = (k < 259) ? in[(size_t)(m0 + m) * 259 + k] : 0.f;
        sInp[i] = (_Float16)v;
    }
    __syncthreads();

    for (int layer = 0; layer < 8; ++layer) {
        const int nK   = c_nK[layer];
        const int fout = c_fout[layer];
        const unsigned wOffB = c_wOff[layer] * 2u;   // byte offset of this layer's fragments

        v8f acc[4][4];
        #pragma unroll
        for (int mt = 0; mt < 4; ++mt)
            #pragma unroll
            for (int ft = 0; ft < 4; ++ft)
                acc[mt][ft] = (v8f)0.f;

        if (layer == 0) {
            gemm_seg(sInp, 288, 0, 9,  wsb, wOffB, nK, wave, ml, kbA, lane, acc);
        } else if (layer == 4) {
            gemm_seg(sAct, 512, 0, 8,  wsb, wOffB, nK, wave, ml, kbA, lane, acc);  // prev act (K 0..255)
            gemm_seg(sInp, 288, 8, 9,  wsb, wOffB, nK, wave, ml, kbA, lane, acc);  // raw input (K 256..543)
        } else {
            gemm_seg(sAct, 512, 0, 16, wsb, wOffB, nK, wave, ml, kbA, lane, acc);
        }
        __syncthreads();   // all reads of sAct complete before in-place overwrite

        // bias + tanh epilogue -> sAct in place (f16); padded F columns written as 0
        const float* bias = P.b[layer];
        #pragma unroll
        for (int mt = 0; mt < 4; ++mt)
            #pragma unroll
            for (int ft = 0; ft < 4; ++ft) {
                const int f = (wave * 4 + ft) * 16 + ml;
                const bool live = (f < fout);
                const float bb = live ? bias[f] : 0.f;
                #pragma unroll
                for (int v = 0; v < 8; ++v) {
                    const int m = mt * 16 + hiRow + v;
                    float t = live ? fast_tanh(acc[mt][ft][v] + bb) : 0.f;
                    sAct[m * 512 + f] = (_Float16)t;
                }
            }
        __syncthreads();

        // LayerNorm over the fout valid features of each row (8 rows per wave)
        const float* gamma = P.g[layer];
        const float* beta  = P.be[layer];
        #pragma unroll
        for (int r = 0; r < 8; ++r) {
            const int m = wave * 8 + r;
            float s1 = 0.f, s2 = 0.f;
            for (int k = lane; k < fout; k += 32) {
                float v = (float)sAct[m * 512 + k];
                s1 += v; s2 += v * v;
            }
            #pragma unroll
            for (int off = 16; off > 0; off >>= 1) {
                s1 += __shfl_xor(s1, off, 32);
                s2 += __shfl_xor(s2, off, 32);
            }
            const float invn = 1.f / (float)fout;
            const float mu = s1 * invn;
            const float var = fmaxf(s2 * invn - mu * mu, 0.f);
            const float rstd = rsqrtf(var + 1e-5f);
            for (int k = lane; k < fout; k += 32) {
                float v = (float)sAct[m * 512 + k];
                sAct[m * 512 + k] = (_Float16)((v - mu) * rstd * gamma[k] + beta[k]);
            }
        }
        __syncthreads();
    }

    // final layer: 512 -> 1, tanh (vectorized VALU dot product, one point per thread)
    if (tid < MTILE) {
        const _Float16* w8 = ws + OFF8;
        const _Float16* row = sAct + tid * 512;
        float accv = 0.f;
        #pragma unroll 4
        for (int k = 0; k < 512; k += 8) {
            h8 r = *(const h8*)(row + k);
            h8 w = *(const h8*)(w8 + k);
            #pragma unroll
            for (int j = 0; j < 8; ++j)
                accv += (float)r[j] * (float)w[j];
        }
        out[m0 + tid] = fast_tanh(accv + P.b[8][0]);
    }
}

// ---------------- host launch ----------------
extern "C" void kernel_launch(void* const* d_in, const int* in_sizes, int n_in,
                              void* d_out, int out_size, void* d_ws, size_t ws_size,
                              hipStream_t stream) {
    (void)in_sizes; (void)n_in; (void)out_size; (void)ws_size;
    // d_in order: input, Ws[0..8], bs[0..8], cs[0..8], gammas[0..7], betas[0..7]
    PrepParams pp;
    DecParams  dp;
    for (int i = 0; i < 9; ++i) {
        pp.W[i] = (const float*)d_in[1 + i];
        pp.c[i] = (const float*)d_in[19 + i];
        dp.b[i] = (const float*)d_in[10 + i];
    }
    for (int i = 0; i < 8; ++i) {
        dp.g[i]  = (const float*)d_in[28 + i];
        dp.be[i] = (const float*)d_in[36 + i];
    }

    // 1) zero weight workspace (fragment + layer-3 F padding must be 0)
    zero_ws_kernel<<<1024, 256, 0, stream>>>((unsigned*)d_ws, WS_HALF_COUNT / 2);
    // 2) Lipschitz-normalize + pack weights to f16 fragments
    prep_weights_kernel<<<9, 256, 0, stream>>>(pp, (_Float16*)d_ws);
    // 3) fused 9-layer decoder, 64 points per block, WMMA f16 GEMMs, in-place LDS activations
    const size_t smem_bytes = (size_t)(MTILE * 288 + MTILE * 512) * sizeof(_Float16); // 102400 B
    decoder_fused_kernel<<<NBLK, THREADS, smem_bytes, stream>>>(
        dp, (const float*)d_in[0], (const _Float16*)d_ws, (float*)d_out);
}
